// ViT_1838246002998
// MI455X (gfx1250) — matrix operand order repaired and checked by reading, not compiled, not run on hardware
//
#include <hip/hip_runtime.h>
#include <math.h>

// ---------------- types ----------------
typedef __attribute__((ext_vector_type(16))) __bf16 v16bf;
typedef __attribute__((ext_vector_type(8)))  float  v8f;

#define DEPTH 6
#define DIM   1024
#define HEADS 8
#define DHEAD 64
#define INNER 512
#define MLP_D 2048
#define SEQ   192
#define NB    16
#define MROWS (NB * SEQ)   // 3072

__device__ __forceinline__ __bf16 f2bf(float f) {
  union { float f; unsigned u; } v; v.f = f;
  unsigned r = v.u + 0x7FFFu + ((v.u >> 16) & 1u);   // round-to-nearest-even
  unsigned short h = (unsigned short)(r >> 16);
  __bf16 out; __builtin_memcpy(&out, &h, 2); return out;
}

// ---------------- preprocess: RGB->YUV, 8x8 DCT, zigzag mask, blockify ----------------
__device__ __forceinline__ int zigzag_index(int i, int j) {
  int d = i + j, off = 0;
  for (int t = 0; t < d; ++t) off += (t < 8) ? (t + 1) : (15 - t);
  int lo = (d < 8) ? 0 : (d - 7);
  int pos = (d & 1) ? ((d < 8) ? (d - j) : (7 - j)) : (j - lo);
  return off + pos;
}

__device__ __forceinline__ void build_dct(float* sD, int t) {
  int k = t >> 3, i = t & 7;
  float d = cosf(3.14159265358979323846f * (float)(2 * i + 1) * (float)k / 16.0f) * 0.5f;
  if (k == 0) d *= 0.70710678118654752f;
  sD[t] = d;
}

__global__ void preprocess_kernel(const float* __restrict__ img, float* __restrict__ x) {
  __shared__ float sD[64];
  __shared__ float sY[3][64];
  __shared__ float sT[3][64];
  const int blk = blockIdx.x;      // hb*32+wb
  const int n   = blockIdx.y;
  const int t   = threadIdx.x;     // 0..63
  const int i = t >> 3, j = t & 7;
  build_dct(sD, t);
  const int hb = blk >> 5, wb = blk & 31;
  const int yy = hb * 8 + i, xx = wb * 8 + j;
  const float* ip = img + (size_t)n * 3 * 65536 + (size_t)yy * 256 + xx;
  float r = ip[0], g = ip[65536], b = ip[2 * 65536];
  sY[0][t] =  0.299f   * r + 0.587f   * g + 0.114f   * b;
  sY[1][t] = -0.14713f * r - 0.28886f * g + 0.436f   * b;
  sY[2][t] =  0.615f   * r - 0.51499f * g - 0.10001f * b;
  __syncthreads();
  #pragma unroll
  for (int c = 0; c < 3; ++c) {
    float acc = 0.f;
    #pragma unroll
    for (int ii = 0; ii < 8; ++ii) acc += sD[i * 8 + ii] * sY[c][ii * 8 + j];
    sT[c][t] = acc;
  }
  __syncthreads();
  const int zz = zigzag_index(i, j);
  #pragma unroll
  for (int c = 0; c < 3; ++c) {
    float acc = 0.f;
    #pragma unroll
    for (int jj = 0; jj < 8; ++jj) acc += sT[c][i * 8 + jj] * sD[j * 8 + jj];
    const int keep = (c == 0) ? 25 : 9;
    if (zz >= keep) acc = 0.f;
    x[((size_t)n * SEQ + c * 64 + t) * DIM + blk] = acc;
  }
}

// ---------------- postprocess: inverse DCT, YUV->RGB ----------------
__global__ void postprocess_kernel(const float* __restrict__ x, float* __restrict__ out) {
  __shared__ float sD[64];
  __shared__ float sC[3][64];
  __shared__ float sU[3][64];
  const int blk = blockIdx.x;
  const int n   = blockIdx.y;
  const int t   = threadIdx.x;
  const int i = t >> 3, j = t & 7;
  build_dct(sD, t);
  #pragma unroll
  for (int c = 0; c < 3; ++c)
    sC[c][t] = x[((size_t)n * SEQ + c * 64 + t) * DIM + blk];
  __syncthreads();
  #pragma unroll
  for (int c = 0; c < 3; ++c) {
    float acc = 0.f;
    #pragma unroll
    for (int q = 0; q < 8; ++q) acc += sD[j * 8 + q] * sC[c][i * 8 + q];
    sU[c][t] = acc;
  }
  __syncthreads();
  float yuv[3];
  #pragma unroll
  for (int c = 0; c < 3; ++c) {
    float acc = 0.f;
    #pragma unroll
    for (int p = 0; p < 8; ++p) acc += sD[i * 8 + p] * sU[c][p * 8 + j];
    yuv[c] = acc;
  }
  const float R = yuv[0] + 1.13983f * yuv[2];
  const float G = yuv[0] - 0.39465f * yuv[1] - 0.5806f * yuv[2];
  const float B = yuv[0] + 2.03211f * yuv[1];
  const int hb = blk >> 5, wb = blk & 31;
  const int yy = hb * 8 + i, xx = wb * 8 + j;
  const size_t base = (size_t)n * 3 * 65536 + (size_t)yy * 256 + xx;
  out[base] = R; out[base + 65536] = G; out[base + 2 * 65536] = B;
}

// ---------------- f32 [K][N] -> bf16 [N][K] tiled transpose ----------------
__global__ void cvt_transpose_bf16(const float* __restrict__ in, __bf16* __restrict__ out,
                                   int K, int N) {
  __shared__ float tile[32][33];
  const int kb = blockIdx.y * 32, nb = blockIdx.x * 32;
  const int tx = threadIdx.x, ty = threadIdx.y;        // 32 x 8
  #pragma unroll
  for (int r = 0; r < 32; r += 8)
    tile[ty + r][tx] = in[(size_t)(kb + ty + r) * N + nb + tx];
  __syncthreads();
  #pragma unroll
  for (int r = 0; r < 32; r += 8)
    out[(size_t)(nb + ty + r) * K + kb + tx] = f2bf(tile[tx][ty + r]);
}

// ---------------- LayerNorm (f32 in -> bf16 out) ----------------
__global__ void ln_kernel(const float* __restrict__ x, const float* __restrict__ g,
                          const float* __restrict__ b, __bf16* __restrict__ out) {
  __shared__ float red[256];
  const int row = blockIdx.x;
  const int tid = threadIdx.x;
  const float* xr = x + (size_t)row * DIM;
  float s = 0.f;
  for (int i = tid; i < DIM; i += 256) s += xr[i];
  red[tid] = s; __syncthreads();
  for (int st = 128; st > 0; st >>= 1) { if (tid < st) red[tid] += red[tid + st]; __syncthreads(); }
  const float mu = red[0] * (1.0f / DIM);
  __syncthreads();
  float v = 0.f;
  for (int i = tid; i < DIM; i += 256) { float d = xr[i] - mu; v += d * d; }
  red[tid] = v; __syncthreads();
  for (int st = 128; st > 0; st >>= 1) { if (tid < st) red[tid] += red[tid + st]; __syncthreads(); }
  const float rstd = rsqrtf(red[0] * (1.0f / DIM) + 1e-5f);
  for (int i = tid; i < DIM; i += 256)
    out[(size_t)row * DIM + i] = f2bf((xr[i] - mu) * rstd * g[i] + b[i]);
}

// ---------------- bf16 WMMA GEMM: C = A[MxK] * Bt[NxK]^T (+epilogue) ----------------
// Workgroup: 256 threads = 8 waves in 2x4; wave tile = 32x64 (2x4 WMMA tiles);
// workgroup tile = 64 x 256.
// B panel (256 rows x 32 K = 16KB) is double-buffer staged into LDS with
// global_load_async_to_lds_b128 (ASYNCcnt pipeline); K-contiguous layout is
// preserved so B fragments are contiguous 32B LDS reads. A fragments are
// K-contiguous in global and loaded directly (L0-resident).
// MODE 0: store bf16 (no bias)          -> outB
// MODE 1: +bias +residual, store f32    -> resOut (ld = N)
// MODE 2: +bias, exact GELU, store bf16 -> outB
template <int MODE>
__global__ void gemm_bf16_kernel(const __bf16* __restrict__ A, const __bf16* __restrict__ Bt,
                                 const float* __restrict__ bias, float* __restrict__ resOut,
                                 __bf16* __restrict__ outB, int M, int N, int K) {
  __shared__ __bf16 Bs[2][256][32];    // 2 x 16KB double buffer
  const int tid  = threadIdx.x;
  const int lane = tid & 31;
  const int w    = tid >> 5;
  const int wm   = w >> 2;               // 0..1
  const int wn   = w & 3;                // 0..3
  const int half = lane >> 4;
  const int ln   = lane & 15;
  const int m0 = blockIdx.y * 64 + wm * 32;
  const int nBlk = blockIdx.x * 256;
  const int n0 = nBlk + wn * 64;

  v8f acc[2][4] = {};

  const __bf16* aRow0 = A + (size_t)(m0 + ln) * K;        // A-frag rows (M = lane&15)
  const __bf16* aRow1 = A + (size_t)(m0 + 16 + ln) * K;

  // async-stage one 256x32 B panel: 1024 16B chunks, 4 per thread
  const int cRow = tid >> 1;              // panel row pair base: rows tid>>1? see mapping below
  (void)cRow;
  auto issue_panel = [&](int k0, int buf) {
    #pragma unroll
    for (int t = 0; t < 4; ++t) {
      const int c   = tid + t * 256;      // chunk id 0..1023
      const int row = c >> 2;             // 0..255
      const int sub = c & 3;              // 16B sub-chunk within 64B row
      const unsigned ldsOff =
          (unsigned)((const char*)&Bs[buf][row][sub * 8] - (const char*)&Bs[0][0][0]);
      const __bf16* g = Bt + (size_t)(nBlk + row) * K + k0 + sub * 8;
      asm volatile("global_load_async_to_lds_b128 %0, %1, off"
                   :: "v"(ldsOff), "v"(g) : "memory");
    }
  };

  const int nsteps = K / 32;
  issue_panel(0, 0);
  for (int s = 0; s < nsteps; ++s) {
    const int buf = s & 1;
    const bool hasNext = (s + 1 < nsteps);
    if (hasNext) issue_panel((s + 1) * 32, buf ^ 1);
    if (hasNext) asm volatile("s_wait_asynccnt 0x4" ::: "memory");
    else         asm volatile("s_wait_asynccnt 0x0" ::: "memory");
    __syncthreads();                      // panel s visible to all waves

    const int k0 = s * 32;
    v16bf af[2], bf[4];
    #pragma unroll
    for (int i = 0; i < 2; ++i) {
      const __bf16* ap = (i ? aRow1 : aRow0) + k0 + half * 16;
      #pragma unroll
      for (int e = 0; e < 8; ++e) { af[i][e] = ap[e]; af[i][e + 8] = ap[8 + e]; }
    }
    #pragma unroll
    for (int j = 0; j < 4; ++j) {
      const __bf16* bp = &Bs[buf][wn * 64 + j * 16 + ln][half * 16];
      #pragma unroll
      for (int e = 0; e < 16; ++e) bf[j][e] = bp[e];
    }
    if (hasNext)  // hint next A chunk -> global_prefetch_b8
      __builtin_prefetch((const void*)(aRow0 + k0 + 32), 0, 1);
    #pragma unroll
    for (int i = 0; i < 2; ++i)
      #pragma unroll
      for (int j = 0; j < 4; ++j)
        acc[i][j] = __builtin_amdgcn_wmma_f32_16x16x32_bf16(
            false, af[i], false, bf[j], (short)0, acc[i][j], false, false);
    __syncthreads();                      // everyone done with buf before it is re-filled
  }

  // epilogue: C/D layout -> m = r + 8*half, n = lane&15
  #pragma unroll
  for (int i = 0; i < 2; ++i) {
    #pragma unroll
    for (int j = 0; j < 4; ++j) {
      const int mt = m0 + i * 16;
      const int nn = n0 + j * 16 + ln;
      #pragma unroll
      for (int r = 0; r < 8; ++r) {
        const int mm = mt + r + half * 8;
        float v = acc[i][j][r];
        if (MODE == 0) {
          outB[(size_t)mm * N + nn] = f2bf(v);
        } else if (MODE == 1) {
          float t = v + bias[nn] + resOut[(size_t)mm * N + nn];
          resOut[(size_t)mm * N + nn] = t;
        } else {
          float t = v + bias[nn];
          t = 0.5f * t * (1.0f + erff(t * 0.70710678118654752f));  // exact GELU
          outB[(size_t)mm * N + nn] = f2bf(t);
        }
      }
    }
  }
}

// ---------------- attention: one block per (q-tile, head, batch) ----------------
// qkv: [3072 x 1536] bf16 rows = b*192+s, cols = {q:0,k:512,v:1024} + h*64 + d
// o  : [3072 x 512]  bf16
__global__ void attn_kernel(const __bf16* __restrict__ qkv, __bf16* __restrict__ o) {
  __shared__ float  sS[16][SEQ];
  __shared__ __bf16 sP[16][SEQ];
  const int qt = blockIdx.x;   // 0..11
  const int h  = blockIdx.y;   // 0..7
  const int bN = blockIdx.z;   // 0..15
  const int tid  = threadIdx.x;        // 0..127 (4 waves)
  const int lane = tid & 31;
  const int w    = tid >> 5;
  const int half = lane >> 4;
  const int ln   = lane & 15;
  const int ldq  = 3 * INNER;          // 1536
  const __bf16* base = qkv + (size_t)bN * SEQ * ldq;

  // scores S = (Q K^T) * scale
  for (int nt = w; nt < 12; nt += 4) {
    v8f acc = {};
    #pragma unroll
    for (int kk = 0; kk < DHEAD; kk += 32) {
      v16bf af, bf;
      #pragma unroll
      for (int e = 0; e < 16; ++e) {
        const int ka = kk + half * 16 + e;
        af[e] = base[(size_t)(qt * 16 + ln) * ldq + h * 64 + ka];            // Q
        const int kb = kk + half * 16 + e;
        bf[e] = base[(size_t)(nt * 16 + ln) * ldq + INNER + h * 64 + kb];    // K
      }
      acc = __builtin_amdgcn_wmma_f32_16x16x32_bf16(false, af, false, bf, (short)0, acc, false, false);
    }
    #pragma unroll
    for (int r = 0; r < 8; ++r)
      sS[r + half * 8][nt * 16 + ln] = acc[r] * 0.125f;  // DHEAD^-0.5
  }
  __syncthreads();

  // softmax per query row
  if (tid < 16) {
    float mx = -1e30f;
    for (int jj = 0; jj < SEQ; ++jj) mx = fmaxf(mx, sS[tid][jj]);
    float sum = 0.f;
    for (int jj = 0; jj < SEQ; ++jj) { float e = expf(sS[tid][jj] - mx); sS[tid][jj] = e; sum += e; }
    const float inv = 1.0f / sum;
    for (int jj = 0; jj < SEQ; ++jj) sP[tid][jj] = f2bf(sS[tid][jj] * inv);
  }
  __syncthreads();

  // O = P V  (wave owns d-tile w)
  {
    v8f acc = {};
    for (int kt = 0; kt < SEQ / 32; ++kt) {
      v16bf af, bf;
      #pragma unroll
      for (int e = 0; e < 16; ++e) {
        const int ka = kt * 32 + half * 16 + e;
        af[e] = sP[ln][ka];
        const int kb = kt * 32 + half * 16 + e;
        bf[e] = base[(size_t)kb * ldq + 2 * INNER + h * 64 + w * 16 + ln];   // V
      }
      acc = __builtin_amdgcn_wmma_f32_16x16x32_bf16(false, af, false, bf, (short)0, acc, false, false);
    }
    #pragma unroll
    for (int r = 0; r < 8; ++r)
      o[((size_t)bN * SEQ + qt * 16 + r + half * 8) * INNER + h * 64 + w * 16 + ln] = f2bf(acc[r]);
  }
}

// ---------------- host launch ----------------
extern "C" void kernel_launch(void* const* d_in, const int* in_sizes, int n_in,
                              void* d_out, int out_size, void* d_ws, size_t ws_size,
                              hipStream_t stream) {
  const float* img   = (const float*)d_in[0];
  const float* ln1_g = (const float*)d_in[1];
  const float* ln1_b = (const float*)d_in[2];
  const float* wqkv  = (const float*)d_in[3];
  const float* wo    = (const float*)d_in[4];
  const float* bo    = (const float*)d_in[5];
  const float* ln2_g = (const float*)d_in[6];
  const float* ln2_b = (const float*)d_in[7];
  const float* w1    = (const float*)d_in[8];
  const float* b1    = (const float*)d_in[9];
  const float* w2    = (const float*)d_in[10];
  const float* b2    = (const float*)d_in[11];
  float* out = (float*)d_out;

  char* ws = (char*)d_ws;
  size_t off = 0;
  auto carve = [&](size_t bytes) { void* p = (void*)(ws + off); off += (bytes + 255) & ~(size_t)255; return p; };

  float*  x    = (float*)carve((size_t)MROWS * DIM * 4);
  __bf16* xn   = (__bf16*)carve((size_t)MROWS * DIM * 2);
  __bf16* qkvB = (__bf16*)carve((size_t)MROWS * 3 * INNER * 2);
  __bf16* oB   = (__bf16*)carve((size_t)MROWS * INNER * 2);
  __bf16* hB   = (__bf16*)carve((size_t)MROWS * MLP_D * 2);
  __bf16* wqkvT = (__bf16*)carve((size_t)DIM * 3 * INNER * 2);   // [1536][1024]
  __bf16* woT   = (__bf16*)carve((size_t)INNER * DIM * 2);       // [1024][512]
  __bf16* w1T   = (__bf16*)carve((size_t)DIM * MLP_D * 2);       // [2048][1024]
  __bf16* w2T   = (__bf16*)carve((size_t)MLP_D * DIM * 2);       // [1024][2048]
  (void)ws_size; (void)in_sizes; (void)n_in; (void)out_size;

  const dim3 tb(32, 8);

  preprocess_kernel<<<dim3(1024, NB), 64, 0, stream>>>(img, x);

  for (int l = 0; l < DEPTH; ++l) {
    const int nqkv = DIM * 3 * INNER, nwo = INNER * DIM, nw1 = DIM * MLP_D, nw2 = MLP_D * DIM;
    cvt_transpose_bf16<<<dim3((3 * INNER) / 32, DIM / 32), tb, 0, stream>>>(
        wqkv + (size_t)l * nqkv, wqkvT, DIM, 3 * INNER);
    cvt_transpose_bf16<<<dim3(DIM / 32, INNER / 32), tb, 0, stream>>>(
        wo + (size_t)l * nwo, woT, INNER, DIM);
    cvt_transpose_bf16<<<dim3(MLP_D / 32, DIM / 32), tb, 0, stream>>>(
        w1 + (size_t)l * nw1, w1T, DIM, MLP_D);
    cvt_transpose_bf16<<<dim3(DIM / 32, MLP_D / 32), tb, 0, stream>>>(
        w2 + (size_t)l * nw2, w2T, MLP_D, DIM);

    // PreNorm attention
    ln_kernel<<<MROWS, 256, 0, stream>>>(x, ln1_g + l * DIM, ln1_b + l * DIM, xn);
    gemm_bf16_kernel<0><<<dim3((3 * INNER) / 256, MROWS / 64), 256, 0, stream>>>(
        xn, wqkvT, nullptr, nullptr, qkvB, MROWS, 3 * INNER, DIM);
    attn_kernel<<<dim3(SEQ / 16, HEADS, NB), 128, 0, stream>>>(qkvB, oB);
    gemm_bf16_kernel<1><<<dim3(DIM / 256, MROWS / 64), 256, 0, stream>>>(
        oB, woT, bo + l * DIM, x, nullptr, MROWS, DIM, INNER);

    // PreNorm MLP
    ln_kernel<<<MROWS, 256, 0, stream>>>(x, ln2_g + l * DIM, ln2_b + l * DIM, xn);
    gemm_bf16_kernel<2><<<dim3(MLP_D / 256, MROWS / 64), 256, 0, stream>>>(
        xn, w1T, b1 + (size_t)l * MLP_D, nullptr, hB, MROWS, MLP_D, DIM);
    gemm_bf16_kernel<1><<<dim3(DIM / 256, MROWS / 64), 256, 0, stream>>>(
        hB, w2T, b2 + l * DIM, x, nullptr, MROWS, DIM, MLP_D);
  }

  postprocess_kernel<<<dim3(1024, NB), 64, 0, stream>>>(x, out);
}
